// BasicRNN_16028817948730
// MI455X (gfx1250) — matrix operand order repaired and checked by reading, not compiled, not run on hardware
//
#include <hip/hip_runtime.h>
#include <hip/hip_bf16.h>
#include <math.h>

// ---------------------------------------------------------------------------
// BasicRNN on MI455X (gfx1250, wave32, WMMA f32_16x16x32_f16)
//   B=256 D=256 T=256 H=512 OUT=256
// Stages:
//   0) pack_w_f16 : fp32 weights -> f16, pre-swizzled into WMMA B-fragment
//      lane order (2x b128 per fragment per lane in the GEMM hot loops)
//   1) xproj_kernel : xproj[t][b][h] = x[b,:,t] @ Wx + b      (WMMA)
//   2) rnn_kernel   : persistent recurrence, 16 WGs (one per 16-batch tile),
//                     h fp32 in registers, f16 copy in LDS as A operand,
//                     Wh f16 streamed from L2 each step          (WMMA)
//   3) out_kernel   : out = states @ Wout + bout                (WMMA)
// ---------------------------------------------------------------------------

typedef __attribute__((ext_vector_type(16))) _Float16 v16h;
typedef __attribute__((ext_vector_type(8)))  _Float16 v8h;
typedef __attribute__((ext_vector_type(8)))  float    v8f;

#define RNN_B   256
#define RNN_D   256
#define RNN_T   256
#define RNN_H   512
#define RNN_OUT 256

// ---------------------------------------------------------------------------
// WMMA wrapper
// ---------------------------------------------------------------------------
__device__ __forceinline__ v8f wmma_f16(v16h a, v16h b, v8f c) {
  return __builtin_amdgcn_wmma_f32_16x16x32_f16(
      /*neg_a=*/false, a, /*neg_b=*/false, b,
      /*c_mod=*/(short)0, c, /*reuse_a=*/false, /*reuse_b=*/false);
}

// A-matrix 16x32 f16 fragment from row-major f16 in LDS.
// ISA 7.12.2: lanes 0-15 hold M=lane, K = k0..k0+7 (VGPR0-3) and
// k0+16..k0+23 (VGPR4-7); lanes 16-31 same rows with K offset +8.
// ld (in halves) must keep 16B alignment: ld*2 % 16 == 0.
__device__ __forceinline__ v16h load_afrag_lds(const _Float16* __restrict__ A,
                                               int ld, int k0, int lane) {
  int row = lane & 15;
  int kb  = k0 + ((lane & 16) ? 8 : 0);
  const _Float16* p = A + row * ld + kb;
  v16h a;
  reinterpret_cast<v8h*>(&a)[0] = *reinterpret_cast<const v8h*>(p);       // K=kb..kb+7
  reinterpret_cast<v8h*>(&a)[1] = *reinterpret_cast<const v8h*>(p + 16);  // K=kb+16..kb+23
  return a;
}

// B-matrix 32x16 f16 fragment from the pre-packed buffer produced by
// pack_w_f16: tile (nt,kt) is 32 lanes x 16 halves contiguous.
__device__ __forceinline__ v16h load_bfrag_packed(const _Float16* __restrict__ P,
                                                  int KT, int nt, int kt, int lane) {
  const _Float16* p = P + ((((size_t)nt * KT + kt) * 32) + lane) * 16;
  v16h b;
  reinterpret_cast<v8h*>(&b)[0] = *reinterpret_cast<const v8h*>(p);
  reinterpret_cast<v8h*>(&b)[1] = *reinterpret_cast<const v8h*>(p + 8);
  return b;
}

// ---------------------------------------------------------------------------
// Stage 0: pack fp32 KxN weight into f16 WMMA B-fragment order.
// Packed element (tile, lane, h):
//   tile = nt*(K/32)+kt, lane: N = nt*16 + (lane&15), K = kt*32 + (lane&16?8:0) + h + (h&8?8:0)
// (matches A fragment: h<8 -> K = koff+h, h>=8 -> K = koff+16+(h-8))
// ---------------------------------------------------------------------------
__global__ __launch_bounds__(256) void pack_w_f16(const float* __restrict__ W,
                                                  _Float16* __restrict__ P,
                                                  int K, int N) {
  int idx = blockIdx.x * 256 + threadIdx.x;
  if (idx >= K * N) return;
  int h    = idx & 15;
  int lane = (idx >> 4) & 31;
  int tile = idx >> 9;
  int KT   = K >> 5;
  int nt   = tile / KT;
  int kt   = tile - nt * KT;
  int n = (nt << 4) + (lane & 15);
  int k = (kt << 5) + ((lane & 16) ? 8 : 0) + h + ((h & 8) ? 8 : 0);
  P[idx] = (_Float16)W[(size_t)k * N + n];
}

// ---------------------------------------------------------------------------
// Stage 1: xproj[t][b][h] = sum_d x[b][d][t] * Wx[d][h] + bias[h]   (fp32 out)
// grid = T * (B/16) workgroups, 256 threads (8 waves), wave owns 4 N-tiles.
// ---------------------------------------------------------------------------
#define XP_LD 264   // 16 rows * 264 halves (528B rows, 16B aligned)
__global__ __launch_bounds__(256) void xproj_kernel(const float* __restrict__ x,
                                                    const _Float16* __restrict__ WxP,
                                                    const float* __restrict__ bias,
                                                    float* __restrict__ xproj) {
  __shared__ _Float16 Ash[16 * XP_LD];
  const int t  = blockIdx.x >> 4;
  const int b0 = (blockIdx.x & 15) << 4;

  // Stage x[b0..b0+15][:][t] -> f16 LDS (K = D = 256)
  for (int idx = threadIdx.x; idx < 16 * RNN_D; idx += 256) {
    int r = idx >> 8, d = idx & 255;
    Ash[r * XP_LD + d] =
        (_Float16)x[(size_t)(b0 + r) * (RNN_D * RNN_T) + (size_t)d * RNN_T + t];
  }
  __syncthreads();

  const int wave = threadIdx.x >> 5;
  const int lane = threadIdx.x & 31;
  const int KT   = RNN_D / 32;   // 8

  v8f acc[4] = {};
#pragma unroll
  for (int kt = 0; kt < KT; ++kt) {
    v16h a = load_afrag_lds(Ash, XP_LD, kt * 32, lane);
#pragma unroll
    for (int j = 0; j < 4; ++j) {
      v16h bf = load_bfrag_packed(WxP, KT, wave * 4 + j, kt, lane);
      acc[j] = wmma_f16(a, bf, acc[j]);
    }
  }

  // C layout: VGPR r -> M = r + (lane>=16 ? 8 : 0), N = lane&15
  const int mh = (lane & 16) ? 8 : 0;
  const int n  = lane & 15;
#pragma unroll
  for (int j = 0; j < 4; ++j) {
    int n0 = (wave * 4 + j) * 16;
    float bv = bias[n0 + n];
    size_t base = ((size_t)t * RNN_B + b0) * RNN_H + n0 + n;
#pragma unroll
    for (int r = 0; r < 8; ++r)
      xproj[base + (size_t)(mh + r) * RNN_H] = acc[j][r] + bv;
  }
}

// ---------------------------------------------------------------------------
// Stage 2: persistent recurrence. 16 workgroups, one per 16-row batch tile.
//   h_new = tanh(xproj_t + h @ Wh) + h
// h kept fp32 in registers (C-fragment layout, 4 tiles/wave), f16 copy in
// LDS as the shared A operand. Wh fragments stream from L2 (512KB resident).
// ---------------------------------------------------------------------------
#define H_LD 520    // 16 rows * 520 halves (1040B rows, 16B aligned)
__global__ __launch_bounds__(256) void rnn_kernel(const _Float16* __restrict__ WhP,
                                                  const float* __restrict__ xproj,
                                                  const float* __restrict__ init_state,
                                                  _Float16* __restrict__ states) {
  __shared__ _Float16 Hs[16 * H_LD];
  const int b0   = blockIdx.x << 4;
  const int wave = threadIdx.x >> 5;
  const int lane = threadIdx.x & 31;
  const int mh   = (lane & 16) ? 8 : 0;
  const int n    = lane & 15;
  const int KT   = RNN_H / 32;   // 16

  // init h (broadcast init_state row) in registers + LDS
  v8f h[4];
#pragma unroll
  for (int j = 0; j < 4; ++j) {
    float hv = init_state[(wave * 4 + j) * 16 + n];
#pragma unroll
    for (int r = 0; r < 8; ++r) h[j][r] = hv;
  }
  for (int idx = threadIdx.x; idx < 16 * RNN_H; idx += 256) {
    int r = idx >> 9, c = idx & 511;
    Hs[r * H_LD + c] = (_Float16)init_state[c];
  }
  __syncthreads();

  for (int t = 0; t < RNN_T; ++t) {
    v8f acc[4] = {};
#pragma unroll 4
    for (int kt = 0; kt < KT; ++kt) {
      v16h a = load_afrag_lds(Hs, H_LD, kt * 32, lane);
      if (kt + 1 < KT)   // hide L2 latency on the serialized critical path
        __builtin_prefetch(WhP + (((size_t)(wave * 4) * KT + kt + 1) * 32) * 16, 0, 1);
#pragma unroll
      for (int j = 0; j < 4; ++j) {
        v16h bf = load_bfrag_packed(WhP, KT, wave * 4 + j, kt, lane);
        acc[j] = wmma_f16(a, bf, acc[j]);
      }
    }
    __syncthreads();   // everyone done reading h_t from LDS

#pragma unroll
    for (int j = 0; j < 4; ++j) {
      int n0 = (wave * 4 + j) * 16;
      size_t xbase = ((size_t)t * RNN_B + b0) * RNN_H + n0 + n;
#pragma unroll
      for (int r = 0; r < 8; ++r) {
        int m = mh + r;
        float nv = tanhf(acc[j][r] + xproj[xbase + (size_t)m * RNN_H]) + h[j][r];
        h[j][r] = nv;
        _Float16 hv = (_Float16)nv;
        Hs[m * H_LD + n0 + n] = hv;                                     // next A
        states[((size_t)(b0 + m) * RNN_T + t) * RNN_H + n0 + n] = hv;   // [b][t][h]
      }
    }
    __syncthreads();   // h_{t+1} visible before next step's A loads
  }
}

// ---------------------------------------------------------------------------
// Stage 3: out[(b,t)][o] = states[(b,t)][:] @ Wout + bout   (M=65536,K=512,N=256)
// grid = M/16 workgroups, A tile staged in LDS, wave owns 2 N-tiles.
// ---------------------------------------------------------------------------
__global__ __launch_bounds__(256) void out_kernel(const _Float16* __restrict__ states,
                                                  const _Float16* __restrict__ WoutP,
                                                  const float* __restrict__ bout,
                                                  float* __restrict__ out) {
  __shared__ _Float16 Ash[16 * H_LD];
  const size_t row0 = (size_t)blockIdx.x * 16;

  // states rows are contiguous f16 (K=512): coalesced v8h copies into LDS
  for (int idx = threadIdx.x; idx < 16 * (RNN_H / 8); idx += 256) {
    int r = idx >> 6, c = idx & 63;
    *reinterpret_cast<v8h*>(Ash + r * H_LD + c * 8) =
        *reinterpret_cast<const v8h*>(states + (row0 + r) * RNN_H + c * 8);
  }
  __syncthreads();

  const int wave = threadIdx.x >> 5;
  const int lane = threadIdx.x & 31;
  const int KT   = RNN_H / 32;   // 16

  v8f acc[2] = {};
#pragma unroll
  for (int kt = 0; kt < KT; ++kt) {
    v16h a = load_afrag_lds(Ash, H_LD, kt * 32, lane);
#pragma unroll
    for (int j = 0; j < 2; ++j) {
      v16h bf = load_bfrag_packed(WoutP, KT, wave * 2 + j, kt, lane);
      acc[j] = wmma_f16(a, bf, acc[j]);
    }
  }

  const int mh = (lane & 16) ? 8 : 0;
  const int n  = lane & 15;
#pragma unroll
  for (int j = 0; j < 2; ++j) {
    int n0 = (wave * 2 + j) * 16;
    float bv = bout[n0 + n];
#pragma unroll
    for (int r = 0; r < 8; ++r)
      out[(row0 + mh + r) * RNN_OUT + n0 + n] = acc[j][r] + bv;
  }
}

// ---------------------------------------------------------------------------
// Host-side launcher. Inputs (setup_inputs order):
//   0:x[B,D,T] 1:Wx[D,H] 2:Wh[H,H] 3:b[H] 4:Wout[H,OUT] 5:bout[OUT] 6:init[1,H]
// Workspace layout (bytes):
//   [0,256K)        WxP  f16 packed
//   [256K,768K)     WhP  f16 packed
//   [768K,1M)       WoutP f16 packed
//   [1M, 1M+128M)   xproj fp32 [T][B][H]
//   [.., +64M)      states f16 [B][T][H]
// ---------------------------------------------------------------------------
extern "C" void kernel_launch(void* const* d_in, const int* in_sizes, int n_in,
                              void* d_out, int out_size, void* d_ws, size_t ws_size,
                              hipStream_t stream) {
  (void)in_sizes; (void)n_in; (void)out_size; (void)ws_size;
  const float* x    = (const float*)d_in[0];
  const float* Wx   = (const float*)d_in[1];
  const float* Wh   = (const float*)d_in[2];
  const float* bias = (const float*)d_in[3];
  const float* Wout = (const float*)d_in[4];
  const float* bout = (const float*)d_in[5];
  const float* init = (const float*)d_in[6];
  float* out = (float*)d_out;

  char* ws = (char*)d_ws;
  _Float16* WxP    = (_Float16*)(ws);
  _Float16* WhP    = (_Float16*)(ws + (256u << 10));
  _Float16* WoutP  = (_Float16*)(ws + (768u << 10));
  float*    xproj  = (float*)   (ws + (1u << 20));
  _Float16* states = (_Float16*)(ws + (1u << 20) + ((size_t)RNN_T * RNN_B * RNN_H * 4));

  pack_w_f16<<<(RNN_D * RNN_H + 255) / 256, 256, 0, stream>>>(Wx, WxP, RNN_D, RNN_H);
  pack_w_f16<<<(RNN_H * RNN_H + 255) / 256, 256, 0, stream>>>(Wh, WhP, RNN_H, RNN_H);
  pack_w_f16<<<(RNN_H * RNN_OUT + 255) / 256, 256, 0, stream>>>(Wout, WoutP, RNN_H, RNN_OUT);

  xproj_kernel<<<RNN_T * (RNN_B / 16), 256, 0, stream>>>(x, WxP, bias, xproj);
  rnn_kernel<<<RNN_B / 16, 256, 0, stream>>>(WhP, xproj, init, states);
  out_kernel<<<(RNN_B * RNN_T) / 16, 256, 0, stream>>>(states, WoutP, bout, out);
}
